// FeedFoward_88261577933469
// MI455X (gfx1250) — matrix operand order repaired and checked by reading, not compiled, not run on hardware
//
#include <hip/hip_runtime.h>
#include <hip/hip_bf16.h>
#include <math.h>

// ---------------------------------------------------------------------------
// BitLinear FFN for MI455X (gfx1250, wave32, WMMA).
//   h   = quant( x @ sign(W1-mean_g)^T + b1 );  h = gelu(h)
//   out = quant( h @ sign(W2-mean_g)^T + b2 )
// Binary weights exact in bf16; activations bf16; fp32 accumulation via
// v_wmma_f32_16x16x32_bf16 with double-buffered fragment loads.
// ---------------------------------------------------------------------------

typedef __attribute__((ext_vector_type(16))) __bf16 v16bf;
typedef __attribute__((ext_vector_type(8)))  __bf16 v8bf;
typedef __attribute__((ext_vector_type(8)))  float  v8f;

#define NUM_GROUPS 5
#define QB 128.0f
#define EPS 1e-5f

// ---------------- weight binarization: one block per group ------------------
__global__ void bl_binarize(const float* __restrict__ W, __bf16* __restrict__ Wb,
                            int n_out, int n_in, int gs) {
  __shared__ float sred[256];
  const int tid = threadIdx.x;
  const int g = blockIdx.x;
  if (g < NUM_GROUPS) {
    const size_t base = (size_t)g * gs * n_in;
    const size_t cnt  = (size_t)gs * n_in;
    float s = 0.0f;
    for (size_t i = tid; i < cnt; i += 256) s += W[base + i];
    sred[tid] = s;
    __syncthreads();
    for (int off = 128; off > 0; off >>= 1) {
      if (tid < off) sred[tid] += sred[tid + off];
      __syncthreads();
    }
    const float mean = sred[0] / (float)cnt;
    for (size_t i = tid; i < cnt; i += 256) {
      float z = W[base + i] - mean;
      float sgn = (z > 0.0f) ? 1.0f : ((z < 0.0f) ? -1.0f : 0.0f);
      Wb[base + i] = (__bf16)sgn;
    }
  } else {
    // tail rows (n_out not divisible by NUM_GROUPS) stay zero
    const size_t start = (size_t)NUM_GROUPS * gs * n_in;
    const size_t end   = (size_t)n_out * n_in;
    for (size_t i = start + tid; i < end; i += 256) Wb[i] = (__bf16)0.0f;
  }
}

// ---------------- fp32 -> bf16 convert --------------------------------------
__global__ void bl_cvt_bf16(const float* __restrict__ src, __bf16* __restrict__ dst,
                            size_t n) {
  size_t stride = (size_t)gridDim.x * blockDim.x;
  for (size_t i = (size_t)blockIdx.x * blockDim.x + threadIdx.x; i < n; i += stride)
    dst[i] = (__bf16)src[i];
}

// ---------------- zero the group-max slots ----------------------------------
__global__ void bl_init_gmax(unsigned* g) {
  if (threadIdx.x < 2 * NUM_GROUPS + 6) g[threadIdx.x] = 0u;
}

// ---------------- fragment loaders (contiguous-k order, same for A and B) ----------------
// A 16x32 bf16: lane = M%16, half h: elems i -> K = k0+h*16+i.
// B 32x16 bf16 (stored as Bt[N,K]): lane = N%16, elems i -> K = k0+h*16+i.
__device__ __forceinline__ void bl_load_frags(const __bf16* __restrict__ A,
                                              const __bf16* __restrict__ Bt,
                                              int m0, int n0, int K, int k0,
                                              int lr, int lh,
                                              v16bf a[4], v16bf b[2]) {
#pragma unroll
  for (int nt = 0; nt < 2; ++nt) {
    const __bf16* bp = Bt + (size_t)(n0 + nt * 16 + lr) * K + k0 + lh * 16;
    v8bf blo = *reinterpret_cast<const v8bf*>(bp);
    v8bf bhi = *reinterpret_cast<const v8bf*>(bp + 8);
#pragma unroll
    for (int i = 0; i < 8; ++i) { b[nt][i] = blo[i]; b[nt][i + 8] = bhi[i]; }
  }
#pragma unroll
  for (int mt = 0; mt < 4; ++mt) {
    const __bf16* ap = A + (size_t)(m0 + mt * 16 + lr) * K + k0 + lh * 16;
    v8bf alo = *reinterpret_cast<const v8bf*>(ap);
    v8bf ahi = *reinterpret_cast<const v8bf*>(ap + 8);
#pragma unroll
    for (int i = 0; i < 8; ++i) { a[mt][i] = alo[i]; a[mt][i + 8] = ahi[i]; }
  }
}

__device__ __forceinline__ void bl_wmma_all(const v16bf a[4], const v16bf b[2],
                                            v8f acc[4][2]) {
#pragma unroll
  for (int mt = 0; mt < 4; ++mt)
#pragma unroll
    for (int nt = 0; nt < 2; ++nt)
      acc[mt][nt] = __builtin_amdgcn_wmma_f32_16x16x32_bf16(
          false, a[mt], false, b[nt], (short)0, acc[mt][nt], false, false);
}

// ---------------- WMMA GEMM:  C[M,N] = A[M,K] * Bt[N,K]^T + bias ------------
// Block: 128 threads = 4 waves (2x2), block tile 128(M) x 64(N),
// wave tile 64(M) x 32(N) = 4x2 tiles of 16x16. Requires M%128==0, N%64==0,
// K%64==0, rows_per_group % 128 == 0. Double-buffered over K.
__global__ void __launch_bounds__(128)
bl_wmma_gemm(const __bf16* __restrict__ A, const __bf16* __restrict__ Bt,
             const float* __restrict__ bias, float* __restrict__ Cout,
             unsigned* __restrict__ gmax, int M, int N, int K,
             int rows_per_group) {
  const int lane = threadIdx.x & 31;
  const int wave = threadIdx.x >> 5;
  const int lr = lane & 15;   // row/col within 16x16 tile
  const int lh = lane >> 4;   // half of wave
  const int m0 = blockIdx.y * 128 + (wave & 1) * 64;
  const int n0 = blockIdx.x * 64 + (wave >> 1) * 32;

  v8f acc[4][2] = {};
  v16bf a0[4], b0[2], a1[4], b1[2];

  // prologue: stage 0 at k=0
  bl_load_frags(A, Bt, m0, n0, K, 0, lr, lh, a0, b0);

  // steady state: branchless, loads for the next stage stay in flight while
  // the current stage's 8 WMMAs execute.
  int k0 = 0;
  for (; k0 + 64 < K; k0 += 64) {
    bl_load_frags(A, Bt, m0, n0, K, k0 + 32, lr, lh, a1, b1);
    bl_wmma_all(a0, b0, acc);
    bl_load_frags(A, Bt, m0, n0, K, k0 + 64, lr, lh, a0, b0);
    bl_wmma_all(a1, b1, acc);
  }
  // epilogue: last 64 of K
  bl_load_frags(A, Bt, m0, n0, K, k0 + 32, lr, lh, a1, b1);
  bl_wmma_all(a0, b0, acc);
  bl_wmma_all(a1, b1, acc);

  // epilogue: bias add, fp32 store, group abs-max
  float lmax = 0.0f;
#pragma unroll
  for (int mt = 0; mt < 4; ++mt) {
#pragma unroll
    for (int nt = 0; nt < 2; ++nt) {
      const int n = n0 + nt * 16 + lr;
      const float bv = bias[n];
#pragma unroll
      for (int r = 0; r < 8; ++r) {
        const int m = m0 + mt * 16 + lh * 8 + r;   // C layout: M = half*8 + r
        float v = acc[mt][nt][r] + bv;
        Cout[(size_t)m * N + n] = v;
        lmax = fmaxf(lmax, fabsf(v));
      }
    }
  }

  __shared__ float sred[128];
  sred[threadIdx.x] = lmax;
  __syncthreads();
  for (int off = 64; off > 0; off >>= 1) {
    if ((int)threadIdx.x < off)
      sred[threadIdx.x] = fmaxf(sred[threadIdx.x], sred[threadIdx.x + off]);
    __syncthreads();
  }
  if (threadIdx.x == 0) {
    const int g = (blockIdx.y * 128) / rows_per_group;  // whole block in one group
    atomicMax(&gmax[g], __float_as_uint(sred[0]));      // values >= 0: uint order ok
  }
}

// ---------------- quantize + exact gelu -> bf16 -----------------------------
// grid.y = activation group; each y-slice handles elems_per_group elements.
__global__ void bl_quant_gelu(const float* __restrict__ y,
                              const unsigned* __restrict__ gmax,
                              __bf16* __restrict__ h,
                              size_t elems_per_group) {
  const size_t base = (size_t)blockIdx.y * elems_per_group;
  const float inv = QB / (__uint_as_float(gmax[blockIdx.y]) + EPS);
  const size_t stride = (size_t)gridDim.x * blockDim.x;
  for (size_t i = (size_t)blockIdx.x * blockDim.x + threadIdx.x;
       i < elems_per_group; i += stride) {
    float q = y[base + i] * inv;
    q = fminf(fmaxf(q, -QB + EPS), QB - EPS);
    const float ge = 0.5f * q * (1.0f + erff(q * 0.70710678118654752f));
    h[base + i] = (__bf16)ge;
  }
}

// ---------------- quantize -> fp32 output -----------------------------------
__global__ void bl_quant_out(const float* __restrict__ y,
                             const unsigned* __restrict__ gmax,
                             float* __restrict__ out,
                             size_t elems_per_group) {
  const size_t base = (size_t)blockIdx.y * elems_per_group;
  const float inv = QB / (__uint_as_float(gmax[blockIdx.y]) + EPS);
  const size_t stride = (size_t)gridDim.x * blockDim.x;
  for (size_t i = (size_t)blockIdx.x * blockDim.x + threadIdx.x;
       i < elems_per_group; i += stride) {
    float q = y[base + i] * inv;
    out[base + i] = fminf(fmaxf(q, -QB + EPS), QB - EPS);
  }
}

// ---------------------------------------------------------------------------
static inline size_t align256(size_t x) { return (x + 255) & ~(size_t)255; }

extern "C" void kernel_launch(void* const* d_in, const int* in_sizes, int n_in,
                              void* d_out, int out_size, void* d_ws, size_t ws_size,
                              hipStream_t stream) {
  const float* x  = (const float*)d_in[0];
  const float* W1 = (const float*)d_in[1];
  const float* b1 = (const float*)d_in[2];
  const float* W2 = (const float*)d_in[3];
  const float* b2 = (const float*)d_in[4];
  float* out = (float*)d_out;

  const int C  = in_sizes[4];                     // 384
  const int H  = in_sizes[2];                     // 1536
  const int BT = in_sizes[0] / C;                 // 20480 rows (B*T)
  const int rpg = BT / NUM_GROUPS;                // 4096 rows per activation group

  // workspace layout (all 256B aligned)
  char* ws = (char*)d_ws;
  size_t off = 0;
  unsigned* gmax = (unsigned*)(ws + off);  off += 256;          // [0..4]=g1, [5..9]=g2
  __bf16* xb  = (__bf16*)(ws + off);  off += align256((size_t)BT * C * 2);
  __bf16* Wb1 = (__bf16*)(ws + off);  off += align256((size_t)H * C * 2);
  __bf16* Wb2 = (__bf16*)(ws + off);  off += align256((size_t)C * H * 2);
  __bf16* hb  = (__bf16*)(ws + off);  off += align256((size_t)BT * H * 2);
  float*  y1  = (float*)(ws + off);   off += align256((size_t)BT * H * 4);
  float*  y2  = y1;                                // y1 dead after bl_quant_gelu
  (void)ws_size; (void)n_in; (void)out_size;

  // 0) reset gammas (deterministic across graph replays)
  bl_init_gmax<<<1, 32, 0, stream>>>(gmax);

  // 1) binarize weights (blocks 0..4 = groups, block 5 = zero tail rows)
  bl_binarize<<<NUM_GROUPS + 1, 256, 0, stream>>>(W1, Wb1, H, C, H / NUM_GROUPS);
  bl_binarize<<<NUM_GROUPS + 1, 256, 0, stream>>>(W2, Wb2, C, H, C / NUM_GROUPS);

  // 2) x -> bf16
  bl_cvt_bf16<<<2048, 256, 0, stream>>>(x, xb, (size_t)BT * C);

  // 3) y1 = xb @ Wb1^T + b1   (M=BT, N=H, K=C), gamma1
  {
    dim3 grid(H / 64, BT / 128);
    bl_wmma_gemm<<<grid, 128, 0, stream>>>(xb, Wb1, b1, y1, gmax, BT, H, C, rpg);
  }

  // 4) h = bf16( gelu( quant(y1) ) )
  {
    dim3 grid(1024, NUM_GROUPS);
    bl_quant_gelu<<<grid, 256, 0, stream>>>(y1, gmax, hb, (size_t)H * rpg);
  }

  // 5) y2 = hb @ Wb2^T + b2   (M=BT, N=C, K=H), gamma2
  {
    dim3 grid(C / 64, BT / 128);
    bl_wmma_gemm<<<grid, 128, 0, stream>>>(hb, Wb2, b2, y2, gmax + NUM_GROUPS,
                                           BT, C, H, rpg);
  }

  // 6) out = quant(y2)
  {
    dim3 grid(1024, NUM_GROUPS);
    bl_quant_out<<<grid, 256, 0, stream>>>(y2, gmax + NUM_GROUPS, out,
                                           (size_t)C * rpg);
  }
}
